// DenseAttention_26345329393804
// MI455X (gfx1250) — compile-verified, matrix operand order checked by reading, not run on hardware
//
#include <hip/hip_runtime.h>
#include <hip/hip_bf16.h>

typedef __attribute__((ext_vector_type(16))) __bf16 v16bf;
typedef __attribute__((ext_vector_type(8)))  float  v8f;
typedef __attribute__((ext_vector_type(4)))  int    v4i;

#define B_  2
#define N_  2048
#define D_  1024
#define NH  16
#define HD  64

#if __has_builtin(__builtin_amdgcn_global_load_async_to_lds_b128) && \
    __has_builtin(__builtin_amdgcn_s_wait_asynccnt)
#define HAS_ASYNC 1
#else
#define HAS_ASYNC 0
#endif

typedef __attribute__((address_space(1))) v4i* gv4i_p;
typedef __attribute__((address_space(3))) v4i* lv4i_p;

// float -> bf16 round-to-nearest-even
__device__ __forceinline__ unsigned short f2bf(float f) {
    unsigned int u = __builtin_bit_cast(unsigned int, f);
    u += 0x7FFFu + ((u >> 16) & 1u);
    return (unsigned short)(u >> 16);
}

union FragU { uint4 u[2]; v16bf v; };

// 16x32 bf16 A-fragment (or mirrored 32x16 B-fragment) from row-major tile,
// leading dim `ld` elements, K contiguous. Lane l: row=l%16, half=l/16;
// VGPR0-3 = K[half*8..+7], VGPR4-7 = K[16+half*8..+7] => two b128 loads.
__device__ __forceinline__ v16bf load_frag(const unsigned short* base, int ld, int lane) {
    int m = lane & 15, half = lane >> 4;
    const unsigned short* p = base + m * ld + half * 8;
    FragU f;
    f.u[0] = *(const uint4*)(p);
    f.u[1] = *(const uint4*)(p + 16);
    return f.v;
}

// 16-byte global -> LDS copy; async (ASYNCcnt) when available.
__device__ __forceinline__ void copy16_g2l(unsigned short* lds_dst, const unsigned short* gsrc) {
#if HAS_ASYNC
    __builtin_amdgcn_global_load_async_to_lds_b128(
        (gv4i_p)(gsrc), (lv4i_p)(lds_dst), 0, 0);
#else
    *(uint4*)lds_dst = *(const uint4*)gsrc;
#endif
}

__device__ __forceinline__ void async_wait0() {
#if HAS_ASYNC
    __builtin_amdgcn_s_wait_asynccnt(0);
#endif
}

// ---------------- conversion kernels ----------------

__global__ __launch_bounds__(256) void cvt_x_kernel(const float* __restrict__ X,
                                                    unsigned short* __restrict__ Xb) {
    int i = (blockIdx.x * 256 + threadIdx.x) * 4;
    float4 f = *(const float4*)(X + i);
    union { unsigned short s[4]; uint2 u; } t;
    t.s[0] = f2bf(f.x); t.s[1] = f2bf(f.y); t.s[2] = f2bf(f.z); t.s[3] = f2bf(f.w);
    *(uint2*)(Xb + i) = t.u;
}

// Wt[n][k] = bf16(W[k][n])
__global__ __launch_bounds__(256) void cvt_wt_kernel(const float* __restrict__ W,
                                                     unsigned short* __restrict__ Wt) {
    __shared__ unsigned short tile[32][33];
    int kt = blockIdx.x * 32, nt = blockIdx.y * 32;
    int c = threadIdx.x & 31, r0 = threadIdx.x >> 5;
#pragma unroll
    for (int i = 0; i < 4; i++) {
        int r = r0 + i * 8;
        tile[r][c] = f2bf(W[(kt + r) * D_ + nt + c]);
    }
    __syncthreads();
#pragma unroll
    for (int i = 0; i < 4; i++) {
        int r = r0 + i * 8;
        Wt[(nt + r) * D_ + kt + c] = tile[c][r];
    }
}

// ---------------- Q projection: Q = X @ W, async double-buffered ----------------
// grid (M/128, Dout/64); 8 waves, wave w owns rows w*16..+15.

__device__ __forceinline__ void proj_stage(unsigned short* As, unsigned short* Bs,
                                           const unsigned short* X, const unsigned short* Wt,
                                           int mb, int nb, int kb, int tid) {
#pragma unroll
    for (int i = 0; i < 4; i++) {                 // A tile: 128x64
        int idx = tid + i * 256;
        int r = idx >> 3, c = (idx & 7) * 8;
        copy16_g2l(As + r * 64 + c, X + (size_t)(mb + r) * D_ + kb + c);
    }
#pragma unroll
    for (int i = 0; i < 2; i++) {                 // B tile: 64(n) x 64(k)
        int idx = tid + i * 256;
        int r = idx >> 3, c = (idx & 7) * 8;
        copy16_g2l(Bs + r * 64 + c, Wt + (size_t)(nb + r) * D_ + kb + c);
    }
}

__global__ __launch_bounds__(256) void proj_kernel(const unsigned short* __restrict__ X,
                                                   const unsigned short* __restrict__ Wt,
                                                   unsigned short* __restrict__ Q) {
    __shared__ __align__(16) unsigned short As[2][128 * 64];
    __shared__ __align__(16) unsigned short Bs[2][64 * 64];
    int tid = threadIdx.x, lane = tid & 31, wave = tid >> 5;
    int mb = blockIdx.x * 128;
    int nb = blockIdx.y * 64;
    v8f acc[4] = {};

    proj_stage(As[0], Bs[0], X, Wt, mb, nb, 0, tid);
    async_wait0();
    __syncthreads();

    const int KSTEPS = D_ / 64;
    for (int ks = 0; ks < KSTEPS; ks++) {
        int cur = ks & 1;
        if (ks + 1 < KSTEPS)
            proj_stage(As[cur ^ 1], Bs[cur ^ 1], X, Wt, mb, nb, (ks + 1) * 64, tid);

        const unsigned short* Ac = As[cur];
        const unsigned short* Bc = Bs[cur];
        v16bf a0 = load_frag(Ac + wave * 16 * 64, 64, lane);
        v16bf a1 = load_frag(Ac + wave * 16 * 64 + 32, 64, lane);
        v16bf b0[4], b1[4];
#pragma unroll
        for (int t = 0; t < 4; t++) {
            b0[t] = load_frag(Bc + t * 16 * 64, 64, lane);
            b1[t] = load_frag(Bc + t * 16 * 64 + 32, 64, lane);
        }
#pragma unroll
        for (int t = 0; t < 4; t++) {
            acc[t] = __builtin_amdgcn_wmma_f32_16x16x32_bf16(false, a0, false, b0[t],
                                                             (short)0, acc[t], false, false);
            acc[t] = __builtin_amdgcn_wmma_f32_16x16x32_bf16(false, a1, false, b1[t],
                                                             (short)0, acc[t], false, false);
        }
        async_wait0();
        __syncthreads();
    }

    int coln = lane & 15, rowb = (lane >> 4) * 8;
#pragma unroll
    for (int t = 0; t < 4; t++)
#pragma unroll
        for (int v = 0; v < 8; v++) {
            int row = mb + wave * 16 + rowb + v;
            int col = nb + t * 16 + coln;
            Q[(size_t)row * D_ + col] = f2bf(acc[t][v]);
        }
}

// ---------------- fused causal DenseAttention: O = tril(Q K^T) V ----------------
// grid (N/128, B*H). K staged async (double-buffered); V^T built LDS->LDS from K
// tile (X read once per block instead of twice).

__global__ __launch_bounds__(256) void attn_kernel(const unsigned short* __restrict__ Xb,
                                                   const unsigned short* __restrict__ Qb,
                                                   float* __restrict__ Out) {
    __shared__ __align__(16) unsigned short Kt[2][64 * 64];   // [m][d]
    __shared__ __align__(16) unsigned short Vt[64 * 64];      // [d][m]
    __shared__ __align__(16) unsigned short Ss[8 * 16 * 64];  // per-wave S tile [16][64]

    int tid = threadIdx.x, lane = tid & 31, wave = tid >> 5;
    int bh = blockIdx.y;
    int b = bh >> 4, h = bh & 15;
    int qb = blockIdx.x * 128;
    int qrow0 = qb + wave * 16;

    const unsigned short* Qbase = Qb + ((size_t)(b * N_ + qrow0)) * D_ + h * HD;
    v16bf q0 = load_frag(Qbase, D_, lane);
    v16bf q1 = load_frag(Qbase + 32, D_, lane);

    v8f o[4] = {};
    const unsigned short* Xh = Xb + (size_t)b * N_ * D_ + h * HD;
    int nkb = (qb >> 6) + 2;
    int coln = lane & 15, rowb = (lane >> 4) * 8;

    // prologue: stage K block 0
#pragma unroll
    for (int i = 0; i < 2; i++) {
        int idx = tid + i * 256;
        int m = idx >> 3, c = (idx & 7) * 8;
        copy16_g2l(Kt[0] + m * 64 + c, Xh + (size_t)m * D_ + c);
    }
    async_wait0();
    __syncthreads();

    for (int j = 0; j < nkb; j++) {
        int kb = j * 64;
        int cur = j & 1;
        const unsigned short* Ktc = Kt[cur];

        // prefetch next K block into the other buffer (overlaps with compute)
        if (j + 1 < nkb) {
#pragma unroll
            for (int i = 0; i < 2; i++) {
                int idx = tid + i * 256;
                int m = idx >> 3, c = (idx & 7) * 8;
                copy16_g2l(Kt[cur ^ 1] + m * 64 + c, Xh + (size_t)(kb + 64 + m) * D_ + c);
            }
        }

        // build V^T [d][m] from resident K tile (LDS -> LDS)
#pragma unroll
        for (int i = 0; i < 2; i++) {
            int idx = tid + i * 256;
            int m = idx >> 3, c = (idx & 7) * 8;
            uint4 val = *(const uint4*)(Ktc + m * 64 + c);
            const unsigned short* pv = (const unsigned short*)&val;
#pragma unroll
            for (int e = 0; e < 8; e++)
                Vt[(c + e) * 64 + m] = pv[e];
        }

        bool active = (kb <= qrow0 + 15);
        if (active) {
            // S = Q * K^T  (batch fragment loads, then WMMAs)
            v16bf kf0[4], kf1[4];
#pragma unroll
            for (int t = 0; t < 4; t++) {
                kf0[t] = load_frag(Ktc + t * 16 * 64, 64, lane);
                kf1[t] = load_frag(Ktc + t * 16 * 64 + 32, 64, lane);
            }
            v8f s[4] = {};
#pragma unroll
            for (int t = 0; t < 4; t++) {
                s[t] = __builtin_amdgcn_wmma_f32_16x16x32_bf16(false, q0, false, kf0[t],
                                                               (short)0, s[t], false, false);
                s[t] = __builtin_amdgcn_wmma_f32_16x16x32_bf16(false, q1, false, kf1[t],
                                                               (short)0, s[t], false, false);
            }
            // causal mask + spill S (bf16) into per-wave LDS staging
            bool need_mask = (kb + 63) > qrow0;
            unsigned short* Sw = Ss + wave * 16 * 64;
#pragma unroll
            for (int t = 0; t < 4; t++)
#pragma unroll
                for (int v = 0; v < 8; v++) {
                    int M = rowb + v;
                    int kidx = kb + t * 16 + coln;
                    float val = s[t][v];
                    if (need_mask && kidx > qrow0 + M) val = 0.0f;
                    Sw[M * 64 + t * 16 + coln] = f2bf(val);
                }
        }
        __syncthreads();   // Vt + Ss ready

        if (active) {
            const unsigned short* Sw = Ss + wave * 16 * 64;
            v16bf sa0 = load_frag(Sw, 64, lane);
            v16bf sa1 = load_frag(Sw + 32, 64, lane);
            v16bf vf0[4], vf1[4];
#pragma unroll
            for (int t = 0; t < 4; t++) {
                vf0[t] = load_frag(Vt + t * 16 * 64, 64, lane);
                vf1[t] = load_frag(Vt + t * 16 * 64 + 32, 64, lane);
            }
#pragma unroll
            for (int t = 0; t < 4; t++) {
                o[t] = __builtin_amdgcn_wmma_f32_16x16x32_bf16(false, sa0, false, vf0[t],
                                                               (short)0, o[t], false, false);
                o[t] = __builtin_amdgcn_wmma_f32_16x16x32_bf16(false, sa1, false, vf1[t],
                                                               (short)0, o[t], false, false);
            }
        }
        async_wait0();     // next K block resident before anyone reads it
        __syncthreads();
    }

    // write O: out[b][n][h*64+d], f32
#pragma unroll
    for (int t = 0; t < 4; t++)
#pragma unroll
        for (int v = 0; v < 8; v++) {
            int n = qrow0 + rowb + v;
            int d = t * 16 + coln;
            Out[((size_t)(b * N_ + n)) * D_ + h * HD + d] = o[t][v];
        }
}

// ---------------- launcher ----------------

extern "C" void kernel_launch(void* const* d_in, const int* in_sizes, int n_in,
                              void* d_out, int out_size, void* d_ws, size_t ws_size,
                              hipStream_t stream) {
    const float* X = (const float*)d_in[0];   // [B,N,D] f32
    const float* W = (const float*)d_in[1];   // [D,D] f32
    float* Out = (float*)d_out;               // [B,N,D] f32

    unsigned short* Xb = (unsigned short*)d_ws;                 // 8 MB
    unsigned short* Wt = Xb + (size_t)B_ * N_ * D_;             // 2 MB
    unsigned short* Qb = Wt + (size_t)D_ * D_;                  // 8 MB

    cvt_x_kernel<<<(B_ * N_ * D_) / (256 * 4), 256, 0, stream>>>(X, Xb);
    cvt_wt_kernel<<<dim3(D_ / 32, D_ / 32), 256, 0, stream>>>(W, Wt);
    proj_kernel<<<dim3((B_ * N_) / 128, D_ / 64), 256, 0, stream>>>(Xb, Wt, Qb);
    attn_kernel<<<dim3(N_ / 128, B_ * NH), 256, 0, stream>>>(Xb, Qb, Out);
}
